// AdreQwen2MLP_50354196578640
// MI455X (gfx1250) — matrix operand order repaired and checked by reading, not compiled, not run on hardware
//
#include <hip/hip_runtime.h>
#include <hip/hip_bf16.h>

// ---------------------------------------------------------------------------
// LoRA-MoE SwiGLU MLP for MI455X (gfx1250, wave32)
//   - bf16 WMMA (v_wmma_f32_16x16x32_bf16) for the three big GEMMs
//   - async global->LDS staging (GLOBAL_LOAD_ASYNC_TO_LDS_B128, ASYNCcnt)
//     with double buffering; WMMA fragments fed from ds_load_b128
//   - rank-16 LoRA deltas as scalar FMAs in the epilogues (~2% of FLOPs)
// ---------------------------------------------------------------------------

typedef __attribute__((ext_vector_type(16))) __bf16       v16bf;
typedef __attribute__((ext_vector_type(8)))  float        v8f;
typedef __attribute__((ext_vector_type(8)))  unsigned int v8u;

// Problem dims fixed by the reference
constexpr int BQ = 4, SQ = 2048, HQ = 2048, IQ = 5504, EQ = 8, RQ = 16;
constexpr float LORA_SCALE = 2.0f;  // 32/16
constexpr int LDK = 40;             // padded LDS row stride in ushorts (80 B)

__device__ __forceinline__ unsigned short f2bf(float f) {
  unsigned u = __builtin_bit_cast(unsigned, f);
  unsigned r = u + 0x7fffu + ((u >> 16) & 1u);   // round-to-nearest-even
  return (unsigned short)(r >> 16);
}
__device__ __forceinline__ float bf2f(unsigned short s) {
  return __builtin_bit_cast(float, (unsigned)s << 16);
}

// one 16-byte async DMA chunk: global -> LDS (tracked by ASYNCcnt)
__device__ __forceinline__ void async_cp16(unsigned lds_off, const unsigned short* g) {
  asm volatile("global_load_async_to_lds_b128 %0, %1, off"
               :: "v"(lds_off), "v"((unsigned long long)(uintptr_t)g)
               : "memory");
}
__device__ __forceinline__ unsigned lds_addr32(const void* p) {
  // LDS aperture: low 32 bits of the generic address are the LDS byte offset
  return (unsigned)(uintptr_t)p;
}

// ---------------------------------------------------------------------------
// 1) top-2 gate selection per batch row (ties -> lower index, like lax.top_k)
// ---------------------------------------------------------------------------
__global__ void topk_kernel(const float* __restrict__ gv, int* __restrict__ eidx) {
  int b = threadIdx.x;
  if (b >= BQ) return;
  const float* row = gv + b * EQ;
  int i1 = 0; float v1 = row[0];
  for (int e = 1; e < EQ; ++e) { float v = row[e]; if (v > v1) { v1 = v; i1 = e; } }
  int i2 = -1; float v2 = -3.402823466e38f;
  for (int e = 0; e < EQ; ++e) {
    if (e == i1) continue;
    float v = row[e]; if (v > v2) { v2 = v; i2 = e; }
  }
  eidx[b * 2 + 0] = i1;
  eidx[b * 2 + 1] = i2;
}

// ---------------------------------------------------------------------------
// 2) f32 -> bf16 conversion
// ---------------------------------------------------------------------------
__global__ void cvt_bf16_kernel(const float* __restrict__ src,
                                unsigned short* __restrict__ dst, int n) {
  int i = blockIdx.x * blockDim.x + threadIdx.x;
  if (i < n) dst[i] = f2bf(src[i]);
}

// ---------------------------------------------------------------------------
// 3) u = x . A_e  for the two selected experts, gate & up adapters (f32)
// ---------------------------------------------------------------------------
__global__ __launch_bounds__(128) void lora_ux_kernel(
    const float* __restrict__ x, const int* __restrict__ eidx,
    const float* __restrict__ Ag, const float* __restrict__ Au,
    float* __restrict__ ug, float* __restrict__ uu) {
  __shared__ float xs[HQ];
  const int token = blockIdx.x;
  const int b = token / SQ;
  for (int k = threadIdx.x; k < HQ; k += blockDim.x)
    xs[k] = x[(size_t)token * HQ + k];
  __syncthreads();
  const int t  = threadIdx.x;
  const int ad = t >> 6;          // 0 = gate adapter, 1 = up adapter
  const int idx = t & 63;
  const int j = idx >> 4;         // expert slot 0/1
  const int r = idx & 15;         // rank index
  const int e = eidx[b * 2 + j];
  const float* A = (ad == 0 ? Ag : Au) + (size_t)e * HQ * RQ + r;
  float s = 0.f;
  #pragma unroll 4
  for (int k = 0; k < HQ; ++k) s += xs[k] * A[(size_t)k * RQ];
  float* u = (ad == 0 ? ug : uu);
  u[token * 32 + j * 16 + r] = s;
}

// ---------------------------------------------------------------------------
// 3b) u_d = h . Ad_e (h is bf16)
// ---------------------------------------------------------------------------
__global__ __launch_bounds__(128) void lora_uh_kernel(
    const unsigned short* __restrict__ h, const int* __restrict__ eidx,
    const float* __restrict__ Ad, float* __restrict__ ud) {
  __shared__ float hsr[IQ];
  const int token = blockIdx.x;
  const int b = token / SQ;
  for (int k = threadIdx.x; k < IQ; k += blockDim.x)
    hsr[k] = bf2f(h[(size_t)token * IQ + k]);
  __syncthreads();
  const int t = threadIdx.x;
  if (t >= 32) return;
  const int j = t >> 4, r = t & 15;
  const int e = eidx[b * 2 + j];
  const float* A = Ad + (size_t)e * IQ * RQ + r;
  float s = 0.f;
  #pragma unroll 4
  for (int k = 0; k < IQ; ++k) s += hsr[k] * A[(size_t)k * RQ];
  ud[token * 32 + j * 16 + r] = s;
}

// ---------------------------------------------------------------------------
// WMMA fragment loaders from LDS-staged tiles (CDNA5 16-bit layouts, ISA 7.12.2)
//   A 16x32: lanes 0-15 hold row M=lane, K in {0..7, 16..23};
//            lanes 16-31 hold K in {8..15, 24..31}  (two 16B ds loads / lane)
//   B 32x16: lane holds col N=lane&15, same K order as A:
//            (lanes 0-15: K in {0..7, 16..23}, lanes 16-31: K in {8..15, 24..31})
// ---------------------------------------------------------------------------
__device__ __forceinline__ v16bf lds_frag_a(const unsigned short* base, int row0, int lane) {
  const int r  = row0 + (lane & 15);
  const int kb = (lane < 16) ? 0 : 8;
  const unsigned short* p = base + r * LDK + kb;
  uint4 lo = *reinterpret_cast<const uint4*>(p);
  uint4 hi = *reinterpret_cast<const uint4*>(p + 16);
  v8u pk;
  pk[0] = lo.x; pk[1] = lo.y; pk[2] = lo.z; pk[3] = lo.w;
  pk[4] = hi.x; pk[5] = hi.y; pk[6] = hi.z; pk[7] = hi.w;
  return __builtin_bit_cast(v16bf, pk);
}
__device__ __forceinline__ v16bf lds_frag_b(const unsigned short* base, int col0, int lane) {
  const int c  = col0 + (lane & 15);
  const int kb = (lane < 16) ? 0 : 8;
  const unsigned short* p = base + c * LDK + kb;
  uint4 lo = *reinterpret_cast<const uint4*>(p);
  uint4 hi = *reinterpret_cast<const uint4*>(p + 16);
  v8u pk;
  pk[0] = lo.x; pk[1] = lo.y; pk[2] = lo.z; pk[3] = lo.w;
  pk[4] = hi.x; pk[5] = hi.y; pk[6] = hi.z; pk[7] = hi.w;
  return __builtin_bit_cast(v16bf, pk);
}

// ---------------------------------------------------------------------------
// 4) fused gate/up GEMM + LoRA delta + SiLU*up -> h (bf16)
//    block = 256 threads (8 waves). Block tile 128(M) x 64(N), K step 32.
//    Double-buffered async staging: X tile + Wg tile + Wu tile per k-step.
//    Each thread DMAs one 64B row slice (4x async b128) -> 4 in flight/wave.
// ---------------------------------------------------------------------------
__global__ __launch_bounds__(256) void gateup_kernel(
    const unsigned short* __restrict__ xb,
    const unsigned short* __restrict__ wgb,
    const unsigned short* __restrict__ wub,
    const float* __restrict__ Bg, const float* __restrict__ Bu,
    const float* __restrict__ ug, const float* __restrict__ uu,
    const int* __restrict__ eidx,
    unsigned short* __restrict__ h) {
  __shared__ __attribute__((aligned(16))) unsigned short xs [2][128 * LDK];
  __shared__ __attribute__((aligned(16))) unsigned short wsg[2][64 * LDK];
  __shared__ __attribute__((aligned(16))) unsigned short wsu[2][64 * LDK];

  const int tid  = threadIdx.x;
  const int lane = tid & 31;
  const int w    = tid >> 5;
  const int m_blk = blockIdx.x * 128;
  const int n_blk = blockIdx.y * 64;
  const int m_w   = (w & 1) * 64;     // wave's local M group (4 subtiles of 16)
  const int n_w   = (w >> 1) * 16;    // wave's local N tile

  auto stage = [&](int buf, int k0) {
    const unsigned short* src;
    unsigned short* dst;
    if (tid < 128)      { src = xb  + (size_t)(m_blk + tid) * HQ + k0;       dst = &xs [buf][tid * LDK]; }
    else if (tid < 192) { src = wgb + (size_t)(n_blk + tid - 128) * HQ + k0; dst = &wsg[buf][(tid - 128) * LDK]; }
    else                { src = wub + (size_t)(n_blk + tid - 192) * HQ + k0; dst = &wsu[buf][(tid - 192) * LDK]; }
    const unsigned dlo = lds_addr32(dst);
    #pragma unroll
    for (int c = 0; c < 4; ++c) async_cp16(dlo + c * 16, src + c * 8);
  };

  v8f cg[4] = {v8f{}, v8f{}, v8f{}, v8f{}};
  v8f cu[4] = {v8f{}, v8f{}, v8f{}, v8f{}};

  stage(0, 0);
  for (int k0 = 0; k0 < HQ; k0 += 32) {
    const int cur = (k0 >> 5) & 1;
    if (k0 + 32 < HQ) {
      stage(cur ^ 1, k0 + 32);
      asm volatile("s_wait_asynccnt 0x4" ::: "memory");  // current stage landed
    } else {
      asm volatile("s_wait_asynccnt 0x0" ::: "memory");
    }
    __syncthreads();
    v16bf bg = lds_frag_b(&wsg[cur][0], n_w, lane);
    v16bf bu = lds_frag_b(&wsu[cur][0], n_w, lane);
    #pragma unroll
    for (int t = 0; t < 4; ++t) {
      v16bf a = lds_frag_a(&xs[cur][0], m_w + t * 16, lane);
      cg[t] = __builtin_amdgcn_wmma_f32_16x16x32_bf16(false, a, false, bg,
                                                      (short)0, cg[t], false, false);
      cu[t] = __builtin_amdgcn_wmma_f32_16x16x32_bf16(false, a, false, bu,
                                                      (short)0, cu[t], false, false);
    }
    __syncthreads();  // all waves done reading `cur` before it is re-staged
  }

  // Epilogue: + LORA_SCALE * (u . B_e), then SiLU(gate) * up, store bf16.
  const int n = n_blk + n_w + (lane & 15);
  const int mhalf = (lane >> 4) << 3;  // lanes 16-31 hold M = i + 8
  #pragma unroll
  for (int t = 0; t < 4; ++t) {
    #pragma unroll
    for (int i = 0; i < 8; ++i) {
      const int m = m_blk + m_w + t * 16 + mhalf + i;
      const int b = m / SQ;
      const float* ugr = ug + m * 32;
      const float* uur = uu + m * 32;
      float dg = 0.f, du = 0.f;
      for (int j = 0; j < 2; ++j) {
        const int e = eidx[b * 2 + j];
        const float* bgrow = Bg + (size_t)(e * RQ) * IQ + n;
        const float* burow = Bu + (size_t)(e * RQ) * IQ + n;
        #pragma unroll
        for (int r = 0; r < RQ; ++r) {
          dg += ugr[j * 16 + r] * bgrow[(size_t)r * IQ];
          du += uur[j * 16 + r] * burow[(size_t)r * IQ];
        }
      }
      const float g  = cg[t][i] + LORA_SCALE * dg;
      const float uv = cu[t][i] + LORA_SCALE * du;
      const float hv = (g / (1.f + __expf(-g))) * uv;  // silu(g) * up
      h[(size_t)m * IQ + n] = f2bf(hv);
    }
  }
}

// ---------------------------------------------------------------------------
// 5) down GEMM + LoRA delta -> out (f32). K = I = 5504, N = H = 2048.
//    Same async double-buffered staging (192 of 256 threads DMA per stage).
// ---------------------------------------------------------------------------
__global__ __launch_bounds__(256) void down_kernel(
    const unsigned short* __restrict__ hb,
    const unsigned short* __restrict__ wdb,
    const float* __restrict__ Bd,
    const float* __restrict__ ud,
    const int* __restrict__ eidx,
    float* __restrict__ out) {
  __shared__ __attribute__((aligned(16))) unsigned short hts[2][128 * LDK];
  __shared__ __attribute__((aligned(16))) unsigned short wds[2][64 * LDK];

  const int tid  = threadIdx.x;
  const int lane = tid & 31;
  const int w    = tid >> 5;
  const int m_blk = blockIdx.x * 128;
  const int n_blk = blockIdx.y * 64;
  const int m_w   = (w & 1) * 64;
  const int n_w   = (w >> 1) * 16;

  auto stage = [&](int buf, int k0) {
    if (tid < 192) {
      const unsigned short* src;
      unsigned short* dst;
      if (tid < 128) { src = hb  + (size_t)(m_blk + tid) * IQ + k0;       dst = &hts[buf][tid * LDK]; }
      else           { src = wdb + (size_t)(n_blk + tid - 128) * IQ + k0; dst = &wds[buf][(tid - 128) * LDK]; }
      const unsigned dlo = lds_addr32(dst);
      #pragma unroll
      for (int c = 0; c < 4; ++c) async_cp16(dlo + c * 16, src + c * 8);
    }
  };

  v8f cd[4] = {v8f{}, v8f{}, v8f{}, v8f{}};

  stage(0, 0);
  for (int k0 = 0; k0 < IQ; k0 += 32) {
    const int cur = (k0 >> 5) & 1;
    if (k0 + 32 < IQ) {
      stage(cur ^ 1, k0 + 32);
      asm volatile("s_wait_asynccnt 0x4" ::: "memory");
    } else {
      asm volatile("s_wait_asynccnt 0x0" ::: "memory");
    }
    __syncthreads();
    v16bf bd = lds_frag_b(&wds[cur][0], n_w, lane);
    #pragma unroll
    for (int t = 0; t < 4; ++t) {
      v16bf a = lds_frag_a(&hts[cur][0], m_w + t * 16, lane);
      cd[t] = __builtin_amdgcn_wmma_f32_16x16x32_bf16(false, a, false, bd,
                                                      (short)0, cd[t], false, false);
    }
    __syncthreads();
  }

  const int n = n_blk + n_w + (lane & 15);
  const int mhalf = (lane >> 4) << 3;
  #pragma unroll
  for (int t = 0; t < 4; ++t) {
    #pragma unroll
    for (int i = 0; i < 8; ++i) {
      const int m = m_blk + m_w + t * 16 + mhalf + i;
      const int b = m / SQ;
      const float* udr = ud + m * 32;
      float dd = 0.f;
      for (int j = 0; j < 2; ++j) {
        const int e = eidx[b * 2 + j];
        const float* bdrow = Bd + (size_t)(e * RQ) * HQ + n;
        #pragma unroll
        for (int r = 0; r < RQ; ++r)
          dd += udr[j * 16 + r] * bdrow[(size_t)r * HQ];
      }
      out[(size_t)m * HQ + n] = cd[t][i] + LORA_SCALE * dd;
    }
  }
}

// ---------------------------------------------------------------------------
// host-side launcher
// ---------------------------------------------------------------------------
extern "C" void kernel_launch(void* const* d_in, const int* in_sizes, int n_in,
                              void* d_out, int out_size, void* d_ws, size_t ws_size,
                              hipStream_t stream) {
  (void)in_sizes; (void)n_in; (void)out_size; (void)ws_size;
  const float* x  = (const float*)d_in[0];
  const float* gv = (const float*)d_in[1];
  const float* Wg = (const float*)d_in[2];
  const float* Ag = (const float*)d_in[3];
  const float* Bg = (const float*)d_in[4];
  const float* Wu = (const float*)d_in[5];
  const float* Au = (const float*)d_in[6];
  const float* Bu = (const float*)d_in[7];
  const float* Wd = (const float*)d_in[8];
  const float* Ad = (const float*)d_in[9];
  const float* Bd = (const float*)d_in[10];
  float* out = (float*)d_out;

  // carve workspace (each region 256B aligned)
  char* ws = (char*)d_ws;
  size_t off = 0;
  auto carve = [&](size_t bytes) -> void* {
    void* p = ws + off;
    off = (off + bytes + 255) & ~(size_t)255;
    return p;
  };
  const size_t ntok = (size_t)BQ * SQ;
  int*            eidx = (int*)carve(BQ * 2 * sizeof(int));
  unsigned short* xb   = (unsigned short*)carve(ntok * HQ * 2);          // x  bf16
  unsigned short* wgb  = (unsigned short*)carve((size_t)IQ * HQ * 2);    // Wg bf16
  unsigned short* wub  = (unsigned short*)carve((size_t)IQ * HQ * 2);    // Wu bf16
  unsigned short* wdb  = (unsigned short*)carve((size_t)HQ * IQ * 2);    // Wd bf16
  unsigned short* hbuf = (unsigned short*)carve(ntok * IQ * 2);          // h  bf16
  float* ug = (float*)carve(ntok * 32 * sizeof(float));
  float* uu = (float*)carve(ntok * 32 * sizeof(float));
  float* ud = (float*)carve(ntok * 32 * sizeof(float));

  topk_kernel<<<1, 32, 0, stream>>>(gv, eidx);

  const int nx = BQ * SQ * HQ;
  const int nw = IQ * HQ;
  cvt_bf16_kernel<<<(nx + 255) / 256, 256, 0, stream>>>(x,  xb,  nx);
  cvt_bf16_kernel<<<(nw + 255) / 256, 256, 0, stream>>>(Wg, wgb, nw);
  cvt_bf16_kernel<<<(nw + 255) / 256, 256, 0, stream>>>(Wu, wub, nw);
  cvt_bf16_kernel<<<(nw + 255) / 256, 256, 0, stream>>>(Wd, wdb, nw);

  lora_ux_kernel<<<(int)ntok, 128, 0, stream>>>(x, eidx, Ag, Au, ug, uu);

  gateup_kernel<<<dim3((unsigned)(ntok / 128), IQ / 64), 256, 0, stream>>>(
      xb, wgb, wub, Bg, Bu, ug, uu, eidx, hbuf);

  lora_uh_kernel<<<(int)ntok, 128, 0, stream>>>(hbuf, eidx, Ad, ud);

  down_kernel<<<dim3((unsigned)(ntok / 128), HQ / 64), 256, 0, stream>>>(
      hbuf, wdb, Bd, ud, eidx, out);
}
